// Model_16690242912472
// MI455X (gfx1250) — matrix operand order repaired and checked by reading, not compiled, not run on hardware
//
#include <hip/hip_runtime.h>
#include <math.h>

typedef _Float16 half16v __attribute__((ext_vector_type(16)));
typedef _Float16 half8v  __attribute__((ext_vector_type(8)));
typedef float    float8v __attribute__((ext_vector_type(8)));
typedef unsigned int uint4v __attribute__((ext_vector_type(4)));
typedef int      int4v  __attribute__((ext_vector_type(4)));
typedef int      int8v  __attribute__((ext_vector_type(8)));

#define LTOT   2048
#define DM     512
#define NHEAD  8
#define DH     64
#define NHASH  4
#define NBUCK  32
#define CS     64
#define NCHUNK 128          // NHASH*NBUCK
#define BATCH  8
#define BH     64           // BATCH*NHEAD
#define MTOT   16384        // BATCH*LTOT
#define DFF    2048
#define SEG    (NHASH*LTOT) // 8192 sorted slots per (b,head)

#if defined(__gfx1250__) && __has_builtin(__builtin_amdgcn_tensor_load_to_lds) && \
    __has_builtin(__builtin_amdgcn_s_wait_tensorcnt)
#define USE_TDM 1
#else
#define USE_TDM 0
#endif

// ---------------------------------------------------------------------------
// WMMA fragment load helpers (16B-aligned LDS vector loads -> ds_load_b128)
// A-frag: lanes 0-15 hold K 0-7 & 16-23, lanes 16-31 hold K 8-15 & 24-31.
// B-frag: same K order as A-frag (loaded via frag_a); frag_b (contiguous 16) is unused.
// ---------------------------------------------------------------------------
__device__ inline half16v frag_a(const _Float16* p0, const _Float16* p1) {
  half8v lo = *(const half8v*)p0;
  half8v hi = *(const half8v*)p1;
  return __builtin_shufflevector(lo, hi, 0,1,2,3,4,5,6,7,8,9,10,11,12,13,14,15);
}
__device__ inline half16v frag_b(const _Float16* p) {
  half8v lo = *(const half8v*)p;
  half8v hi = *(const half8v*)(p + 8);
  return __builtin_shufflevector(lo, hi, 0,1,2,3,4,5,6,7,8,9,10,11,12,13,14,15);
}

#if USE_TDM
// ---------------------------------------------------------------------------
// Tensor Data Mover: async DMA of a rows x 32(half) f16 tile into LDS.
// D# per ISA 8.3/8.4: count=1, type=2, data_size=1(2B), pad 16B per 64B row
// -> LDS row stride = 80B (16B aligned for b128 fragment loads).
// ---------------------------------------------------------------------------
__device__ inline void tdm_tile_rx32_f16(const _Float16* src, unsigned lds_off,
                                         int row_stride_elems, int rows) {
  unsigned long long ga = (unsigned long long)(uintptr_t)src;
  uint4v g0 = { 1u,                                   // count=1 (valid descriptor)
                lds_off,                               // lds_addr
                (unsigned)ga,                          // global_addr[31:0]
                (unsigned)((ga >> 32) & 0x01FFFFFFu) | (2u << 30) }; // ga[56:32] | type=2
  int8v g1 = { (int)((1u << 16) |                     // data_size = 2 bytes
                     (1u << 20) |                     // pad_enable
                     (3u << 22) |                     // pad_interval: 16 DWORDs (64B)
                     (3u << 25)),                     // pad_amount:   4 DWORDs (16B)
               (int)(32u << 16),                      // tensor_dim0 = 32
               (int)((unsigned)rows << 16),           // tensor_dim1 = rows
               (int)(32u << 16),                      // tile_dim0 = 32
               rows,                                  // tile_dim1 = rows
               row_stride_elems,                      // tensor_dim0_stride
               0, 0 };
  int4v z4 = { 0, 0, 0, 0 };
#if defined(__clang_major__) && __clang_major__ >= 23
  int8v z8 = { 0, 0, 0, 0, 0, 0, 0, 0 };
  __builtin_amdgcn_tensor_load_to_lds(g0, g1, z4, z4, z8, 0);
#else
  __builtin_amdgcn_tensor_load_to_lds(g0, g1, z4, z4, 0);
#endif
}
#endif

// ---------------------------------------------------------------------------
// Embedding: circular conv1d(k=3) + sinusoidal PE + time embedding.
// Dual output: f32 (residual stream) + f16 (GEMM operand).
// ---------------------------------------------------------------------------
__global__ void embed_kernel(const float* __restrict__ xe, const float* __restrict__ xme,
                             const float* __restrict__ xd, const float* __restrict__ xmd,
                             const float* __restrict__ convw, const float* __restrict__ timew,
                             float* __restrict__ h, _Float16* __restrict__ h16) {
  int idx = blockIdx.x * blockDim.x + threadIdx.x;    // (b*L + l)*DM + d
  if (idx >= MTOT * DM) return;
  int d  = idx & (DM - 1);
  int bl = idx / DM;
  int l  = bl & (LTOT - 1);
  int b  = bl / LTOT;
  float acc = 0.f;
  #pragma unroll
  for (int t = 0; t < 3; ++t) {
    int ls = (l - 1 + t + LTOT) & (LTOT - 1);
    #pragma unroll
    for (int c = 0; c < 7; ++c) {
      float xv = (ls < 1536) ? xe[(size_t)(b * 1536 + ls) * 7 + c]
                             : xd[(size_t)(b * 1024 + 512 + (ls - 1536)) * 7 + c];
      acc += xv * convw[(size_t)(t * 7 + c) * DM + d];
    }
  }
  int de = d & ~1;
  float freq = expf(-logf(10000.f) * (float)de / (float)DM);
  float ang  = (float)l * freq;
  acc += (d & 1) ? cosf(ang) : sinf(ang);
  #pragma unroll
  for (int f = 0; f < 4; ++f) {
    float mv = (l < 1536) ? xme[(size_t)(b * 1536 + l) * 4 + f]
                          : xmd[(size_t)(b * 1024 + 512 + (l - 1536)) * 4 + f];
    acc += mv * timew[(size_t)f * DM + d];
  }
  h[idx] = acc;
  h16[idx] = (_Float16)acc;
}

// ---------------------------------------------------------------------------
// Weight convert + transpose: WT[n][k] = f16(W[k][n]) (done once per call)
// ---------------------------------------------------------------------------
__global__ void wt_cvt_kernel(const float* __restrict__ W, _Float16* __restrict__ WT,
                              int K, int N) {
  int idx = blockIdx.x * blockDim.x + threadIdx.x;
  if (idx >= K * N) return;
  int k = idx % K, n = idx / K;
  WT[(size_t)n * K + k] = (_Float16)W[(size_t)k * N + n];
}

// ---------------------------------------------------------------------------
// WMMA GEMM: C[M,N] = A[M,K] @ WT[N,K]^T (+bias, optional exact GELU)
// f16 operands, f32 accumulate. 128x64 tile/block, 4 waves; each wave does a
// 32x64 sub-tile (2 A-frags x 4 B-frags -> 8 WMMAs per k-step; B-frag reuse 2x).
// Tiles staged via TDM (double-buffered, s_wait_tensorcnt) when available.
// ---------------------------------------------------------------------------
__global__ void gemm_wmma_kernel(const _Float16* __restrict__ A, const _Float16* __restrict__ WT,
                                 const float* __restrict__ bias, float* __restrict__ Cf,
                                 _Float16* __restrict__ Ch, int M, int K, int N, int fuse_gelu) {
  __shared__ _Float16 As[2][128][40] __attribute__((aligned(16)));  // 80B row stride
  __shared__ _Float16 Bs[2][64][40]  __attribute__((aligned(16)));
  int tid = threadIdx.x;
  int wv = tid >> 5, lane = tid & 31, laneN = lane & 15, hi = lane >> 4;
  int gm0 = blockIdx.y * 128, gn0 = blockIdx.x * 64;
  const _Float16* Ab = A  + (size_t)gm0 * K;
  const _Float16* Bb = WT + (size_t)gn0 * K;

  auto stage = [&](int buf, int k0) {
#if USE_TDM
    if (tid < 32) {
      tdm_tile_rx32_f16(Ab + k0, (unsigned)(uintptr_t)&As[buf][0][0], K, 128);
      tdm_tile_rx32_f16(Bb + k0, (unsigned)(uintptr_t)&Bs[buf][0][0], K, 64);
    }
#else
    for (int i = tid; i < 512; i += 128) {
      int r = i >> 2, c8 = (i & 3) * 8;
      *(half8v*)&As[buf][r][c8] = *(const half8v*)&Ab[(size_t)r * K + k0 + c8];
    }
    for (int i = tid; i < 256; i += 128) {
      int r = i >> 2, c8 = (i & 3) * 8;
      *(half8v*)&Bs[buf][r][c8] = *(const half8v*)&Bb[(size_t)r * K + k0 + c8];
    }
#endif
  };
  auto settle = [&]() {
#if USE_TDM
    if (tid < 32) __builtin_amdgcn_s_wait_tensorcnt(0);
#endif
    __syncthreads();
  };

  float8v acc[2][4] = {};
  stage(0, 0);
  settle();
  int cur = 0;
  for (int k0 = 0; k0 < K; k0 += 32) {
    int nxt = cur ^ 1;
    if (k0 + 32 < K) stage(nxt, k0 + 32);
    int am0 = wv * 32 + laneN;
    half16v af0 = frag_a(&As[cur][am0][hi * 8],      &As[cur][am0][16 + hi * 8]);
    half16v af1 = frag_a(&As[cur][am0 + 16][hi * 8], &As[cur][am0 + 16][16 + hi * 8]);
    #pragma unroll
    for (int nt = 0; nt < 4; ++nt) {
      half16v bf = frag_a(&Bs[cur][nt * 16 + laneN][hi * 8], &Bs[cur][nt * 16 + laneN][16 + hi * 8]);
      acc[0][nt] = __builtin_amdgcn_wmma_f32_16x16x32_f16(false, af0, false, bf,
                                                          (short)0, acc[0][nt], false, false);
      acc[1][nt] = __builtin_amdgcn_wmma_f32_16x16x32_f16(false, af1, false, bf,
                                                          (short)0, acc[1][nt], false, false);
    }
    settle();
    cur = nxt;
  }
  #pragma unroll
  for (int mt = 0; mt < 2; ++mt) {
    #pragma unroll
    for (int nt = 0; nt < 4; ++nt) {
      #pragma unroll
      for (int r = 0; r < 8; ++r) {
        int m = gm0 + wv * 32 + mt * 16 + hi * 8 + r;
        int n = gn0 + nt * 16 + laneN;
        float v = acc[mt][nt][r];
        if (bias) v += bias[n];
        if (fuse_gelu) v = 0.5f * v * (1.f + erff(v * 0.70710678118654752f));
        if (Cf) Cf[(size_t)m * N + n] = v;
        if (Ch) Ch[(size_t)m * N + n] = (_Float16)v;
      }
    }
  }
}

// ---------------------------------------------------------------------------
// LSH hashing: buckets[hb][hash][l] = argmax over [r, -r] of rotated qk
// ---------------------------------------------------------------------------
__global__ void lsh_hash_kernel(const float* __restrict__ QK, const float* __restrict__ rot,
                                int* __restrict__ buckets) {
  int idx = blockIdx.x * blockDim.x + threadIdx.x;   // (hb*L + l)*NHASH + hash
  if (idx >= BH * LTOT * NHASH) return;
  int hash = idx & 3;
  int t  = idx >> 2;
  int l  = t & (LTOT - 1);
  int hb = t >> 11;
  int bb = hb >> 3, hd = hb & 7;
  const float4* q4 = (const float4*)(QK + (size_t)(bb * LTOT + l) * DM + hd * DH);
  float r[16];
  #pragma unroll
  for (int j = 0; j < 16; ++j) r[j] = 0.f;
  for (int d4 = 0; d4 < 16; ++d4) {
    float4 q = q4[d4];
    #pragma unroll
    for (int s = 0; s < 4; ++s) {
      int d = d4 * 4 + s;
      float qd = (s == 0) ? q.x : (s == 1) ? q.y : (s == 2) ? q.z : q.w;
      const float* rr = rot + (size_t)(d * NHASH + hash) * 16;
      #pragma unroll
      for (int j = 0; j < 16; ++j) r[j] += qd * rr[j];
    }
  }
  float best = -1e30f; int bi = 0;
  #pragma unroll
  for (int j = 0; j < 32; ++j) {
    float v = (j < 16) ? r[j] : -r[j - 16];
    if (v > best) { best = v; bi = j; }
  }
  buckets[((size_t)hb * NHASH + hash) * LTOT + l] = bi;
}

// ---------------------------------------------------------------------------
// Stable counting sort per (hb, hash): key = (bucket, position)
// ---------------------------------------------------------------------------
__global__ void lsh_sort_kernel(const int* __restrict__ buckets,
                                int* __restrict__ st, int* __restrict__ undo) {
  __shared__ int lb[LTOT];
  __shared__ int cnt[NBUCK];
  __shared__ int off[NBUCK];
  int g = blockIdx.x;                      // hb*NHASH + hash
  int tid = threadIdx.x;                   // 64 threads
  const int* src = buckets + (size_t)g * LTOT;
  if (tid < NBUCK) cnt[tid] = 0;
  __syncthreads();
  for (int i = tid; i < LTOT; i += 64) { int bv = src[i]; lb[i] = bv; atomicAdd(&cnt[bv], 1); }
  __syncthreads();
  if (tid == 0) { int s = 0; for (int j = 0; j < NBUCK; ++j) { off[j] = s; s += cnt[j]; } }
  __syncthreads();
  if (tid < NBUCK) {
    int pos = off[tid];
    int hb = g >> 2, hash = g & 3;
    size_t base = (size_t)hb * SEG + (size_t)hash * LTOT;
    for (int i = 0; i < LTOT; ++i) {
      if (lb[i] == tid) { st[base + pos] = i; undo[base + i] = pos; ++pos; }
    }
  }
}

// ---------------------------------------------------------------------------
// Chunked LSH attention (per (hb, chunk)): gather, normalize keys, WMMA dots,
// self-mask, row softmax, WMMA p@V. Writes so + logsumexp.
// ---------------------------------------------------------------------------
__global__ void lsh_attn_kernel(const float* __restrict__ QK, const float* __restrict__ V,
                                const int* __restrict__ st, float* __restrict__ so,
                                float* __restrict__ slog) {
  __shared__ _Float16 sQ[64][DH]    __attribute__((aligned(16)));
  __shared__ _Float16 sK[128][DH]   __attribute__((aligned(16)));
  __shared__ _Float16 sVT[DH][128]  __attribute__((aligned(16)));
  __shared__ _Float16 sP[64][128]   __attribute__((aligned(16)));
  __shared__ int tq[64];
  __shared__ int tk[128];
  int c = blockIdx.x, hb = blockIdx.y;
  int bb = hb >> 3, hd = hb & 7;
  int tid = threadIdx.x;                   // 128
  int wv = tid >> 5, lane = tid & 31, laneN = lane & 15, hi = lane >> 4;
  const int* stb = st + (size_t)hb * SEG;

  if (tid < 64) tq[tid] = stb[c * CS + tid];
  {
    int j = tid;
    int cc = (j < 64) ? c : (c + NCHUNK - 1) % NCHUNK;
    tk[j] = stb[cc * CS + (j & 63)];
  }
  __syncthreads();
  if (tid < 64) {
    int l = tq[tid];
    const float4* q4 = (const float4*)(QK + (size_t)(bb * LTOT + l) * DM + hd * DH);
    #pragma unroll 4
    for (int d4 = 0; d4 < 16; ++d4) {
      float4 x = q4[d4];
      sQ[tid][d4*4+0] = (_Float16)x.x; sQ[tid][d4*4+1] = (_Float16)x.y;
      sQ[tid][d4*4+2] = (_Float16)x.z; sQ[tid][d4*4+3] = (_Float16)x.w;
    }
  }
  {
    int j = tid;
    int l = tk[j];
    const float4* q4 = (const float4*)(QK + (size_t)(bb * LTOT + l) * DM + hd * DH);
    const float4* v4 = (const float4*)(V  + (size_t)(bb * LTOT + l) * DM + hd * DH);
    float ss = 0.f;
    #pragma unroll 4
    for (int d4 = 0; d4 < 16; ++d4) {
      float4 x = q4[d4];
      ss += x.x*x.x + x.y*x.y + x.z*x.z + x.w*x.w;
    }
    float rn = 1.f / fmaxf(sqrtf(ss), 1e-12f);
    #pragma unroll 4
    for (int d4 = 0; d4 < 16; ++d4) {
      float4 x = q4[d4];
      sK[j][d4*4+0] = (_Float16)(x.x*rn); sK[j][d4*4+1] = (_Float16)(x.y*rn);
      sK[j][d4*4+2] = (_Float16)(x.z*rn); sK[j][d4*4+3] = (_Float16)(x.w*rn);
    }
    #pragma unroll 4
    for (int d4 = 0; d4 < 16; ++d4) {
      float4 x = v4[d4];
      sVT[d4*4+0][j] = (_Float16)x.x; sVT[d4*4+1][j] = (_Float16)x.y;
      sVT[d4*4+2][j] = (_Float16)x.z; sVT[d4*4+3][j] = (_Float16)x.w;
    }
  }
  __syncthreads();

  // dots: each wave -> 16 rows x 128 cols
  float8v acc[8] = {};
  #pragma unroll
  for (int k0 = 0; k0 < DH; k0 += 32) {
    int am = wv * 16 + laneN;
    half16v af = frag_a(&sQ[am][k0 + hi * 8], &sQ[am][k0 + 16 + hi * 8]);
    #pragma unroll
    for (int nt = 0; nt < 8; ++nt) {
      half16v bf = frag_a(&sK[nt * 16 + laneN][k0 + hi * 8], &sK[nt * 16 + laneN][k0 + 16 + hi * 8]);
      acc[nt] = __builtin_amdgcn_wmma_f32_16x16x32_f16(false, af, false, bf,
                                                       (short)0, acc[nt], false, false);
    }
  }

  // mask + row softmax (row lives in one 16-lane half across 8 N-tiles)
  const float scale = 0.125f;  // d^-0.5, d=64
  #pragma unroll
  for (int r = 0; r < 8; ++r) {
    int row = wv * 16 + hi * 8 + r;
    int myq = tq[row];
    float vals[8];
    float mx = -1e30f;
    #pragma unroll
    for (int nt = 0; nt < 8; ++nt) {
      int n = nt * 16 + laneN;
      float v = acc[nt][r] * scale;
      if (tk[n] == myq) v = -5e4f;
      vals[nt] = v;
      mx = fmaxf(mx, v);
    }
    for (int m_ = 1; m_ <= 8; m_ <<= 1) mx = fmaxf(mx, __shfl_xor(mx, m_, 32));
    float sum = 0.f;
    #pragma unroll
    for (int nt = 0; nt < 8; ++nt) sum += expf(vals[nt] - mx);
    for (int m_ = 1; m_ <= 8; m_ <<= 1) sum += __shfl_xor(sum, m_, 32);
    float lse = mx + logf(sum);
    #pragma unroll
    for (int nt = 0; nt < 8; ++nt) sP[row][nt * 16 + laneN] = (_Float16)expf(vals[nt] - lse);
    if (laneN == 0) slog[(size_t)hb * SEG + c * CS + row] = lse;
  }
  __syncthreads();

  // so = p @ v : 16 x 64 per wave, K=128
  float8v oacc[4] = {};
  #pragma unroll
  for (int k0 = 0; k0 < 128; k0 += 32) {
    int am = wv * 16 + laneN;
    half16v af = frag_a(&sP[am][k0 + hi * 8], &sP[am][k0 + 16 + hi * 8]);
    #pragma unroll
    for (int nt = 0; nt < 4; ++nt) {
      half16v bf = frag_a(&sVT[nt * 16 + laneN][k0 + hi * 8], &sVT[nt * 16 + laneN][k0 + 16 + hi * 8]);
      oacc[nt] = __builtin_amdgcn_wmma_f32_16x16x32_f16(false, af, false, bf,
                                                        (short)0, oacc[nt], false, false);
    }
  }
  #pragma unroll
  for (int nt = 0; nt < 4; ++nt) {
    #pragma unroll
    for (int r = 0; r < 8; ++r) {
      int row = wv * 16 + hi * 8 + r;
      so[((size_t)hb * SEG + c * CS + row) * DH + nt * 16 + laneN] = oacc[nt][r];
    }
  }
}

// ---------------------------------------------------------------------------
// Un-sort + combine across hashes -> f16 (B, L, 512) (feeds out-proj GEMM)
// ---------------------------------------------------------------------------
__global__ void lsh_combine_kernel(const float* __restrict__ so, const float* __restrict__ slog,
                                   const int* __restrict__ undo, _Float16* __restrict__ out16) {
  int idx = blockIdx.x * blockDim.x + threadIdx.x;   // (hb*L + l)*DH + dh
  if (idx >= BH * LTOT * DH) return;
  int dh = idx & (DH - 1);
  int t  = idx >> 6;
  int l  = t & (LTOT - 1);
  int hb = t >> 11;
  int bb = hb >> 3, hd = hb & 7;
  float lg[NHASH], ov[NHASH];
  float mx = -1e30f;
  #pragma unroll
  for (int h = 0; h < NHASH; ++h) {
    int pos = undo[(size_t)hb * SEG + h * LTOT + l];
    size_t s = (size_t)hb * SEG + h * LTOT + pos;
    lg[h] = slog[s];
    ov[h] = so[s * DH + dh];
    mx = fmaxf(mx, lg[h]);
  }
  float den = 0.f;
  #pragma unroll
  for (int h = 0; h < NHASH; ++h) den += expf(lg[h] - mx);
  float o = 0.f;
  #pragma unroll
  for (int h = 0; h < NHASH; ++h) o += ov[h] * (expf(lg[h] - mx) / den);
  out16[(size_t)(bb * LTOT + l) * DM + hd * DH + dh] = (_Float16)o;
}

// ---------------------------------------------------------------------------
// Residual add + LayerNorm; dual output f32 (+optional f16 for GEMMs)
// ---------------------------------------------------------------------------
__global__ void add_ln_kernel(const float* __restrict__ x, const float* __restrict__ r,
                              const float* __restrict__ g, const float* __restrict__ b,
                              float* __restrict__ out, _Float16* __restrict__ out16) {
  int row  = blockIdx.x * 8 + (threadIdx.x >> 5);
  int lane = threadIdx.x & 31;
  const float* xr = x + (size_t)row * DM;
  const float* rr = r ? r + (size_t)row * DM : nullptr;
  float v[16];
  float s = 0.f;
  #pragma unroll
  for (int i = 0; i < 16; ++i) {
    int d = lane + i * 32;
    float t = xr[d] + (rr ? rr[d] : 0.f);
    v[i] = t; s += t;
  }
  for (int m_ = 1; m_ < 32; m_ <<= 1) s += __shfl_xor(s, m_, 32);
  float mu = s * (1.f / DM);
  float q = 0.f;
  #pragma unroll
  for (int i = 0; i < 16; ++i) { float d_ = v[i] - mu; q += d_ * d_; }
  for (int m_ = 1; m_ < 32; m_ <<= 1) q += __shfl_xor(q, m_, 32);
  float rstd = rsqrtf(q * (1.f / DM) + 1e-5f);
  #pragma unroll
  for (int i = 0; i < 16; ++i) {
    int d = lane + i * 32;
    float o = (v[i] - mu) * rstd * g[d] + b[d];
    out[(size_t)row * DM + d] = o;
    if (out16) out16[(size_t)row * DM + d] = (_Float16)o;
  }
}

// ---------------------------------------------------------------------------
// Final projection (N=7, last 512 timesteps)
// ---------------------------------------------------------------------------
__global__ void proj_kernel(const float* __restrict__ hn, const float* __restrict__ pw,
                            const float* __restrict__ pb, float* __restrict__ out) {
  int idx = blockIdx.x * blockDim.x + threadIdx.x;   // (b*512 + lp)*7 + c
  if (idx >= BATCH * 512 * 7) return;
  int cc = idx % 7;
  int t  = idx / 7;
  int lp = t & 511;
  int b  = t >> 9;
  const float* hr = hn + (size_t)(b * LTOT + 1536 + lp) * DM;
  float s = pb[cc];
  for (int d = 0; d < DM; ++d) s += hr[d] * pw[(size_t)d * 7 + cc];
  out[idx] = s;
}

// ---------------------------------------------------------------------------
extern "C" void kernel_launch(void* const* d_in, const int* in_sizes, int n_in,
                              void* d_out, int out_size, void* d_ws, size_t ws_size,
                              hipStream_t stream) {
  const float* x_enc      = (const float*)d_in[0];
  const float* x_mark_enc = (const float*)d_in[1];
  const float* x_dec      = (const float*)d_in[2];
  const float* x_mark_dec = (const float*)d_in[3];
  const float* conv_w     = (const float*)d_in[4];
  const float* time_w     = (const float*)d_in[5];
  const float* qk_w       = (const float*)d_in[6];
  const float* v_w        = (const float*)d_in[7];
  const float* out_w      = (const float*)d_in[8];
  const float* out_b      = (const float*)d_in[9];
  const float* rotations  = (const float*)d_in[10];
  const float* ffn1_w     = (const float*)d_in[11];
  const float* ffn1_b     = (const float*)d_in[12];
  const float* ffn2_w     = (const float*)d_in[13];
  const float* ffn2_b     = (const float*)d_in[14];
  const float* norm1_g    = (const float*)d_in[15];
  const float* norm1_b    = (const float*)d_in[16];
  const float* norm2_g    = (const float*)d_in[17];
  const float* norm2_b    = (const float*)d_in[18];
  const float* normf_g    = (const float*)d_in[19];
  const float* normf_b    = (const float*)d_in[20];
  const float* proj_w     = (const float*)d_in[21];
  const float* proj_b     = (const float*)d_in[22];
  float* outp = (float*)d_out;

  char* p = (char*)d_ws;
  auto carve = [&](size_t bytes) -> void* {
    void* r = (void*)p;
    p += (bytes + 255) & ~(size_t)255;
    return r;
  };
  float*     h    = (float*)carve((size_t)MTOT * DM * 4);
  float*     bufA = (float*)carve((size_t)MTOT * DM * 4);   // QK / gemm f32 outputs
  float*     bufB = (float*)carve((size_t)MTOT * DM * 4);   // V
  float*     big  = (float*)carve((size_t)MTOT * DFF * 4);  // so (sorted attention out)
  float*     slog = (float*)carve((size_t)BH * SEG * 4);
  _Float16*  h16  = (_Float16*)carve((size_t)MTOT * DM * 2);
  _Float16*  c16  = (_Float16*)carve((size_t)MTOT * DM * 2);
  _Float16*  mid16= (_Float16*)carve((size_t)MTOT * DFF * 2);
  _Float16*  wqk16= (_Float16*)carve((size_t)2 * DM * DM * 2);
  _Float16*  wv16 = (_Float16*)carve((size_t)2 * DM * DM * 2);
  _Float16*  wo16 = (_Float16*)carve((size_t)2 * DM * DM * 2);
  _Float16*  wf116= (_Float16*)carve((size_t)2 * DM * DFF * 2);
  _Float16*  wf216= (_Float16*)carve((size_t)2 * DFF * DM * 2);
  int* buckets = (int*)carve((size_t)BH * SEG * 4);
  int* stb     = (int*)carve((size_t)BH * SEG * 4);
  int* undo    = (int*)carve((size_t)BH * SEG * 4);
  (void)in_sizes; (void)n_in; (void)out_size; (void)ws_size;

  // one-time: weights -> f16, transposed to [N][K]
  for (int l = 0; l < 2; ++l) {
    wt_cvt_kernel<<<(DM * DM) / 256, 256, 0, stream>>>(qk_w + (size_t)l * DM * DM,
                                                       wqk16 + (size_t)l * DM * DM, DM, DM);
    wt_cvt_kernel<<<(DM * DM) / 256, 256, 0, stream>>>(v_w + (size_t)l * DM * DM,
                                                       wv16 + (size_t)l * DM * DM, DM, DM);
    wt_cvt_kernel<<<(DM * DM) / 256, 256, 0, stream>>>(out_w + (size_t)l * DM * DM,
                                                       wo16 + (size_t)l * DM * DM, DM, DM);
    wt_cvt_kernel<<<(DM * DFF) / 256, 256, 0, stream>>>(ffn1_w + (size_t)l * DM * DFF,
                                                        wf116 + (size_t)l * DM * DFF, DM, DFF);
    wt_cvt_kernel<<<(DFF * DM) / 256, 256, 0, stream>>>(ffn2_w + (size_t)l * DFF * DM,
                                                        wf216 + (size_t)l * DFF * DM, DFF, DM);
  }

  embed_kernel<<<(MTOT * DM) / 256, 256, 0, stream>>>(x_enc, x_mark_enc, x_dec, x_mark_dec,
                                                      conv_w, time_w, h, h16);

  for (int l = 0; l < 2; ++l) {
    const float* rot = rotations + (size_t)l * DH * NHASH * (NBUCK / 2);

    gemm_wmma_kernel<<<dim3(DM / 64, MTOT / 128), 128, 0, stream>>>(
        h16, wqk16 + (size_t)l * DM * DM, nullptr, bufA, nullptr, MTOT, DM, DM, 0);
    gemm_wmma_kernel<<<dim3(DM / 64, MTOT / 128), 128, 0, stream>>>(
        h16, wv16 + (size_t)l * DM * DM, nullptr, bufB, nullptr, MTOT, DM, DM, 0);
    lsh_hash_kernel<<<(BH * LTOT * NHASH) / 256, 256, 0, stream>>>(bufA, rot, buckets);
    lsh_sort_kernel<<<BH * NHASH, 64, 0, stream>>>(buckets, stb, undo);
    lsh_attn_kernel<<<dim3(NCHUNK, BH), 128, 0, stream>>>(bufA, bufB, stb, big, slog);
    lsh_combine_kernel<<<(BH * LTOT * DH) / 256, 256, 0, stream>>>(big, slog, undo, c16);
    gemm_wmma_kernel<<<dim3(DM / 64, MTOT / 128), 128, 0, stream>>>(
        c16, wo16 + (size_t)l * DM * DM, out_b + (size_t)l * DM, bufA, nullptr,
        MTOT, DM, DM, 0);
    add_ln_kernel<<<MTOT / 8, 256, 0, stream>>>(h, bufA, norm1_g + (size_t)l * DM,
                                                norm1_b + (size_t)l * DM, h, h16);
    gemm_wmma_kernel<<<dim3(DFF / 64, MTOT / 128), 128, 0, stream>>>(
        h16, wf116 + (size_t)l * DM * DFF, ffn1_b + (size_t)l * DFF, nullptr, mid16,
        MTOT, DM, DFF, 1);
    gemm_wmma_kernel<<<dim3(DM / 64, MTOT / 128), 128, 0, stream>>>(
        mid16, wf216 + (size_t)l * DFF * DM, ffn2_b + (size_t)l * DM, bufA, nullptr,
        MTOT, DFF, DM, 0);
    add_ln_kernel<<<MTOT / 8, 256, 0, stream>>>(h, bufA, norm2_g + (size_t)l * DM,
                                                norm2_b + (size_t)l * DM, h, h16);
  }

  add_ln_kernel<<<MTOT / 8, 256, 0, stream>>>(h, nullptr, normf_g, normf_b, bufA, nullptr);
  proj_kernel<<<(BATCH * 512 * 7 + 255) / 256, 256, 0, stream>>>(bufA, proj_w, proj_b, outp);
}
